// GoEModel_74199855006306
// MI455X (gfx1250) — compile-verified, matrix operand order checked by reading, not run on hardware
//
#include <hip/hip_runtime.h>
#include <hip/hip_bf16.h>

#define VOCAB 32000
#define DM    512
#define NEXP  8
#define NH    8
#define FFD   2048
#define RHD   256
#define BB    16
#define SS    256
#define NL    2
#define DH    64

typedef __attribute__((ext_vector_type(16))) __bf16 v16bf;
typedef __attribute__((ext_vector_type(8)))  __bf16 v8bf;
typedef __attribute__((ext_vector_type(8)))  float  v8f;
typedef __attribute__((__vector_size__(16))) int    b128v;   // builtin's payload type

#if __has_builtin(__builtin_amdgcn_global_load_async_to_lds_b128) && \
    __has_builtin(__builtin_amdgcn_s_wait_asynccnt)
#define USE_ASYNC 1
#else
#define USE_ASYNC 0
#endif

#if USE_ASYNC
__device__ __forceinline__ void async_ld_b128(const void* gp, void* lp) {
    __builtin_amdgcn_global_load_async_to_lds_b128(
        (__attribute__((address_space(1))) b128v*)gp,
        (__attribute__((address_space(3))) b128v*)lp, 0, 0);
}
#endif

// ---------- bf16 WMMA fragment loader ----------
// 16-bit A layout (16x32 MxK): lanes 0-15 hold K {0..7,16..23} of row M=lane;
// lanes 16-31 hold K {8..15,24..31}. B (from W[N,K] rows) mirrors with N.
// In bf16 each half is one contiguous 16-byte chunk -> two b128 loads.
__device__ __forceinline__ v16bf frag_bf(const __bf16* rowp, int k0, int lane) {
    int kb = k0 + ((lane & 16) ? 8 : 0);
    v8bf lo = *(const v8bf*)(rowp + kb);
    v8bf hi = *(const v8bf*)(rowp + kb + 16);
    return __builtin_shufflevector(lo, hi, 0, 1, 2, 3, 4, 5, 6, 7,
                                   8, 9, 10, 11, 12, 13, 14, 15);
}

__device__ __forceinline__ v8f wmma_bf16(v16bf a, v16bf b, v8f c) {
    return __builtin_amdgcn_wmma_f32_16x16x32_bf16(false, a, false, b, (short)0, c,
                                                   false, false);
}

// ---------- block GEMM: 128 threads (4 waves), 128x64 tile, 32x64 per wave ----------
// C[M,N] = A[M,K] @ W[N,K]^T (+bias). B tile double-buffered in LDS via async DMA.
template<int RELU, int WF32, int WBF>
__global__ __launch_bounds__(128)
void k_bgemm(const __bf16* A, int lda, long aStr,
             const __bf16* Wb, long wStr, const float* biasB, long bStr,
             float* C, __bf16* Cbf, int ldc, long cStr, int K, int layer,
             const int* choice, const int* cont) {
    __shared__ __align__(16) __bf16 lb[2][64 * 32];
    int bz = blockIdx.z;
    if (cont && !cont[bz]) return;
    long widx = cont ? ((long)choice[bz] * NL + layer) : 0;
    const __bf16* Wt = Wb + widx * wStr;
    const float* bias = biasB ? (biasB + widx * bStr) : nullptr;
    const __bf16* Ab = A + (long)bz * aStr;
    int tM = blockIdx.x * 128;
    int tN = blockIdx.y * 64;
    int t = threadIdx.x;
    int lane = t & 31;
    int wv = t >> 5;
    int wM = tM + wv * 32;
    int KT = K / 32;

    const __bf16* arow0 = Ab + (long)(wM + (lane & 15)) * lda;
    const __bf16* arow1 = arow0 + 16L * lda;

    // stage K-tile 0 into buffer 0 (64 rows x 32 K = 4KB = 256 16B chunks)
#pragma unroll
    for (int i = 0; i < 2; ++i) {
        int c = t + i * 128;
        const __bf16* gp = Wt + (long)(tN + (c >> 2)) * K + (c & 3) * 8;
        __bf16* lp = &lb[0][c * 8];
#if USE_ASYNC
        async_ld_b128(gp, lp);
#else
        *(v8bf*)lp = *(const v8bf*)gp;
#endif
    }

    v8f z = {};
    v8f acc[8];
#pragma unroll
    for (int i = 0; i < 8; ++i) acc[i] = z;

    for (int kt = 0; kt < KT; ++kt) {
        if (kt + 1 < KT) {
            int kb = (kt + 1) * 32;
            int buf = (kt + 1) & 1;
#pragma unroll
            for (int i = 0; i < 2; ++i) {
                int c = t + i * 128;
                const __bf16* gp = Wt + (long)(tN + (c >> 2)) * K + kb + (c & 3) * 8;
                __bf16* lp = &lb[buf][c * 8];
#if USE_ASYNC
                async_ld_b128(gp, lp);
#else
                *(v8bf*)lp = *(const v8bf*)gp;
#endif
            }
#if USE_ASYNC
            __builtin_amdgcn_s_wait_asynccnt(2);   // current stage done, prefetch in flight
#endif
        } else {
#if USE_ASYNC
            __builtin_amdgcn_s_wait_asynccnt(0);
#endif
        }
        __syncthreads();
        int k0 = kt * 32;
        v16bf aF0 = frag_bf(arow0, k0, lane);
        v16bf aF1 = frag_bf(arow1, k0, lane);
        const __bf16* lbase = lb[kt & 1] + (lane & 15) * 32;
#pragma unroll
        for (int j = 0; j < 4; ++j) {
            v16bf bF = frag_bf(lbase + j * (16 * 32), 0, lane);
            acc[j]     = wmma_bf16(aF0, bF, acc[j]);
            acc[4 + j] = wmma_bf16(aF1, bF, acc[4 + j]);
        }
        __syncthreads();   // before prefetch overwrites this buffer
    }

#pragma unroll
    for (int half = 0; half < 2; ++half) {
#pragma unroll
        for (int j = 0; j < 4; ++j) {
            int n = tN + j * 16 + (lane & 15);
            int m0 = wM + half * 16 + ((lane & 16) ? 8 : 0);
            float bv = bias ? bias[n] : 0.f;
            const v8f& a = acc[half * 4 + j];
#pragma unroll
            for (int r = 0; r < 8; ++r) {
                float v = a[r] + bv;
                if (RELU) v = fmaxf(v, 0.f);
                long ci = (long)(m0 + r) * ldc + n;
                if (WF32) (C + (long)bz * cStr)[ci] = v;
                if (WBF)  (Cbf + (long)bz * cStr)[ci] = (__bf16)v;
            }
        }
    }
}

// ---------- attention GEMMs: wave-level, direct bf16 loads ----------
__global__ __launch_bounds__(32)
void k_scores(const __bf16* qkvbf, float* scores, const int* cont) {
    int bh = blockIdx.z, b = bh >> 3, h = bh & 7;
    if (!cont[b]) return;
    int lane = threadIdx.x;
    const __bf16* base = qkvbf + (long)b * SS * (3 * DM);
    int tM = blockIdx.x * 16, tN = blockIdx.y * 64;
    const __bf16* arow = base + h * DH + (long)(tM + (lane & 15)) * (3 * DM);
    v8f zz = {};
    v8f acc[4] = {zz, zz, zz, zz};
    for (int k0 = 0; k0 < DH; k0 += 32) {
        v16bf aF = frag_bf(arow, k0, lane);
#pragma unroll
        for (int j = 0; j < 4; ++j) {
            const __bf16* brow = base + DM + h * DH +
                                 (long)(tN + j * 16 + (lane & 15)) * (3 * DM);
            v16bf bF = frag_bf(brow, k0, lane);
            acc[j] = wmma_bf16(aF, bF, acc[j]);
        }
    }
    float* Cp = scores + (long)bh * SS * SS;
#pragma unroll
    for (int j = 0; j < 4; ++j) {
        int n = tN + j * 16 + (lane & 15);
        int m0 = tM + ((lane & 16) ? 8 : 0);
#pragma unroll
        for (int r = 0; r < 8; ++r) Cp[(long)(m0 + r) * SS + n] = acc[j][r];
    }
}

__global__ __launch_bounds__(32)
void k_attnout(const __bf16* pbf, const __bf16* vt, __bf16* obf, const int* cont) {
    int bh = blockIdx.z, b = bh >> 3, h = bh & 7;
    if (!cont[b]) return;
    int lane = threadIdx.x;
    int tM = blockIdx.x * 16;
    const __bf16* A = pbf + (long)bh * SS * SS;
    const __bf16* W = vt + (long)bh * DH * SS;   // V^T rows: [dh, S]
    const __bf16* arow = A + (long)(tM + (lane & 15)) * SS;
    v8f zz = {};
    v8f acc[4] = {zz, zz, zz, zz};
    for (int k0 = 0; k0 < SS; k0 += 32) {
        v16bf aF = frag_bf(arow, k0, lane);
#pragma unroll
        for (int j = 0; j < 4; ++j) {
            v16bf bF = frag_bf(W + (long)(j * 16 + (lane & 15)) * SS, k0, lane);
            acc[j] = wmma_bf16(aF, bF, acc[j]);
        }
    }
    __bf16* Cp = obf + (long)b * SS * DM + h * DH;
#pragma unroll
    for (int j = 0; j < 4; ++j) {
        int n = j * 16 + (lane & 15);
        int m0 = tM + ((lane & 16) ? 8 : 0);
#pragma unroll
        for (int r = 0; r < 8; ++r) Cp[(long)(m0 + r) * DM + n] = (__bf16)acc[j][r];
    }
}

// ---------- elementwise / small kernels ----------
__global__ void k_cvt(const float* src, __bf16* dst, long n) {
    long i = (long)blockIdx.x * 256 + threadIdx.x;
    if (i < n) dst[i] = (__bf16)src[i];
}

__global__ void k_vt(const __bf16* qkvbf, __bf16* vt, const int* cont) {
    long i = (long)blockIdx.x * 256 + threadIdx.x;   // B*H*DH*SS
    if (i >= (long)BB * NH * DH * SS) return;
    int b = (int)(i / ((long)NH * DH * SS));
    if (!cont[b]) return;
    int h = (int)((i / ((long)DH * SS)) % NH);
    int d = (int)((i / SS) % DH);
    int s = (int)(i % SS);
    vt[i] = qkvbf[((long)b * SS + s) * (3 * DM) + 2 * DM + h * DH + d];
}

__global__ void k_init(int* visits, int* active, float* acc) {
    int t = threadIdx.x;
    if (t < BB * NEXP) visits[t] = 0;
    if (t < BB) active[t] = 1;
    if (t < 2) acc[t] = 0.f;
}

__global__ void k_embed(const int* ids, const float* emb, float* reps,
                        __bf16* reps_bf, __bf16* fin_bf) {
    long i = (long)blockIdx.x * 256 + threadIdx.x;
    if (i >= (long)BB * SS * DM) return;
    int d0 = (int)(i % DM);
    long bs = i / DM;
    int s = (int)(bs % SS);
    int id = ids[bs];
    float v = emb[(long)id * DM + d0] * 22.62741699796952f;  // sqrt(512)
    int half = d0 >> 1;
    float div = expf(-(float)(2 * half) * (logf(10000.f) / (float)DM));
    float ang = (float)s * div;
    v += (d0 & 1) ? cosf(ang) : sinf(ang);
    reps[i] = v;
    reps_bf[i] = (__bf16)v;
    fin_bf[i] = (__bf16)v;
}

__global__ void k_summary(const float* reps, float* summary) {
    int i = blockIdx.x * 256 + threadIdx.x;   // B*D
    if (i >= BB * DM) return;
    int b = i / DM, d0 = i % DM;
    const float* p = reps + (long)b * SS * DM + d0;
    float s = 0.f;
    for (int t = 0; t < SS; ++t) s += p[(long)t * DM];
    summary[i] = s * (1.f / (float)SS);
}

__global__ void k_rfc1(const float* summ, const float* w, const float* bias, float* h) {
    int i = blockIdx.x * 256 + threadIdx.x;   // B*RH
    if (i >= BB * RHD) return;
    int b = i / RHD, r = i % RHD;
    const float* sp = summ + (long)b * DM;
    const float* wp = w + (long)r * DM;
    float a = bias[r];
    for (int d0 = 0; d0 < DM; ++d0) a += sp[d0] * wp[d0];
    h[i] = fmaxf(a, 0.f);
}

__global__ void k_rfc2(const float* h, const float* w, const float* bias, float* lg) {
    int i = blockIdx.x * 192 + threadIdx.x;   // B*(E+1) = 144
    if (i >= BB * (NEXP + 1)) return;
    int b = i / (NEXP + 1), j = i % (NEXP + 1);
    const float* hp = h + (long)b * RHD;
    const float* wp = w + (long)j * RHD;
    float a = bias[j];
    for (int r = 0; r < RHD; ++r) a += hp[r] * wp[r];
    lg[i] = a;
}

__global__ void k_route(const float* lg, const float* g, const float* bb,
                        int* visits, int* active, int* cont, int* choice, int* copyflag,
                        float* ent_acc, float* div_acc) {
    __shared__ float s_probs[BB][NEXP + 1];
    __shared__ float s_ent[BB];
    __shared__ int   s_act[BB];
    int b = threadIdx.x;
    if (b < BB) {
        float x[NEXP + 1];
        float m = 0.f;
        for (int j = 0; j <= NEXP; ++j) { x[j] = lg[b * (NEXP + 1) + j]; m += x[j]; }
        m /= (float)(NEXP + 1);
        float var = 0.f;
        for (int j = 0; j <= NEXP; ++j) { float d = x[j] - m; var += d * d; }
        var /= (float)(NEXP + 1);
        float inv = rsqrtf(var + 1e-5f);
        for (int j = 0; j <= NEXP; ++j) x[j] = (x[j] - m) * inv * g[j] + bb[j];
        for (int e = 0; e < NEXP; ++e)
            if (visits[b * NEXP + e] >= 2) x[e] = -__builtin_inff();
        float mx = x[0]; int am = 0;
        for (int j = 1; j <= NEXP; ++j) if (x[j] > mx) { mx = x[j]; am = j; }
        float p[NEXP + 1], sum = 0.f;
        for (int j = 0; j <= NEXP; ++j) { p[j] = expf(x[j] - mx); sum += p[j]; }
        float ent = 0.f;
        for (int j = 0; j <= NEXP; ++j) {
            float q = p[j] / sum;
            q = fminf(fmaxf(q, 1e-10f), 1.f - 1e-10f);
            p[j] = q;
            ent -= q * logf(q + 1e-10f);
        }
        int act = active[b];
        int term = (am == NEXP);
        copyflag[b] = (act && term) ? 1 : 0;
        int cn = (act && !term) ? 1 : 0;
        if (cn && am < NEXP) visits[b * NEXP + am] += 1;
        cont[b] = cn;
        choice[b] = am;
        s_act[b] = act;
        s_ent[b] = ent * (float)act;
        for (int j = 0; j <= NEXP; ++j) s_probs[b][j] = p[j] * (float)act;
    }
    __syncthreads();
    if (threadIdx.x == 0) {
        int n = 0;
        for (int i = 0; i < BB; ++i) n += s_act[i];
        if (n > 0) {
            float den = (float)n;
            float se = 0.f;
            for (int i = 0; i < BB; ++i) se += s_ent[i];
            *ent_acc += se / den;
            float dv = 0.f;
            for (int j = 0; j <= NEXP; ++j) {
                float ap = 0.f;
                for (int i = 0; i < BB; ++i) ap += s_probs[i][j];
                ap /= den;
                dv -= ap * logf(ap + 1e-10f);
            }
            *div_acc += dv;
        }
    }
    __syncthreads();
    if (b < BB) active[b] = cont[b];
}

__global__ void k_copyfinal(const __bf16* reps_bf, __bf16* fin_bf, const int* flag) {
    long i = (long)blockIdx.x * 256 + threadIdx.x;
    if (i >= (long)BB * SS * DM) return;
    int b = (int)(i / ((long)SS * DM));
    if (!flag[b]) return;
    fin_bf[i] = reps_bf[i];
}

__global__ void k_softmax(const float* scores, __bf16* p_bf, const int* cont) {
    int row = blockIdx.x;                  // B*H*S rows
    int b = row >> 11;                     // / (NH*SS)
    if (!cont[b]) return;
    const float* p = scores + (long)row * SS;
    int t = threadIdx.x;                   // 256
    float v = p[t] * 0.125f;               // 1/sqrt(64)
    __shared__ float red[SS];
    red[t] = v;
    __syncthreads();
    for (int o = 128; o > 0; o >>= 1) { if (t < o) red[t] = fmaxf(red[t], red[t + o]); __syncthreads(); }
    float mx = red[0];
    __syncthreads();
    float e = expf(v - mx);
    red[t] = e;
    __syncthreads();
    for (int o = 128; o > 0; o >>= 1) { if (t < o) red[t] += red[t + o]; __syncthreads(); }
    p_bf[(long)row * SS + t] = (__bf16)(e / red[0]);
}

__global__ void k_resid_ln(const float* x, const float* y, const float* gbase, const float* bbase,
                           float* outp, __bf16* outbf, const int* choice, const int* cont,
                           int layer) {
    int tok = blockIdx.x;                  // B*S tokens
    int b = tok >> 8;
    if (!cont[b]) return;
    long eidx = (long)choice[b] * NL + layer;
    const float* g  = gbase + eidx * DM;
    const float* bi = bbase + eidx * DM;
    const float* xp = x + (long)tok * DM;
    const float* yp = y + (long)tok * DM;
    int t = threadIdx.x;                   // 256 threads, 2 elems each
    float a0 = xp[t] + yp[t];
    float a1 = xp[t + 256] + yp[t + 256];
    __shared__ float rs[256], rq[256];
    rs[t] = a0 + a1;
    rq[t] = a0 * a0 + a1 * a1;
    __syncthreads();
    for (int o = 128; o > 0; o >>= 1) {
        if (t < o) { rs[t] += rs[t + o]; rq[t] += rq[t + o]; }
        __syncthreads();
    }
    float mean = rs[0] * (1.f / (float)DM);
    float var  = rq[0] * (1.f / (float)DM) - mean * mean;
    float inv = rsqrtf(var + 1e-5f);
    float o0 = (a0 - mean) * inv * g[t] + bi[t];
    float o1 = (a1 - mean) * inv * g[t + 256] + bi[t + 256];
    outp[(long)tok * DM + t] = o0;
    outp[(long)tok * DM + t + 256] = o1;
    outbf[(long)tok * DM + t] = (__bf16)o0;
    outbf[(long)tok * DM + t + 256] = (__bf16)o1;
}

__global__ void k_apply(const float* xout, const float* tag, const int* choice, const int* cont,
                        float* reps, __bf16* reps_bf) {
    long i = (long)blockIdx.x * 256 + threadIdx.x;
    if (i >= (long)BB * SS * DM) return;
    int b = (int)(i / ((long)SS * DM));
    if (!cont[b]) return;
    int d0 = (int)(i % DM);
    float v = xout[i] + tag[(long)choice[b] * DM + d0];
    reps[i] = v;
    reps_bf[i] = (__bf16)v;
}

__global__ void k_finalize(const float* acc, float* outp) {
    outp[0] = acc[0];
    outp[1] = acc[1];
}

// ---------- host launch ----------
extern "C" void kernel_launch(void* const* d_in, const int* in_sizes, int n_in,
                              void* d_out, int out_size, void* d_ws, size_t ws_size,
                              hipStream_t stream) {
    const int*   ids   = (const int*)d_in[0];
    const float* emb   = (const float*)d_in[1];
    const float* qkv_w = (const float*)d_in[2];
    const float* qkv_b = (const float*)d_in[3];
    const float* out_w = (const float*)d_in[4];
    const float* out_b = (const float*)d_in[5];
    const float* ln1_g = (const float*)d_in[6];
    const float* ln1_b = (const float*)d_in[7];
    const float* ff1_w = (const float*)d_in[8];
    const float* ff1_b = (const float*)d_in[9];
    const float* ff2_w = (const float*)d_in[10];
    const float* ff2_b = (const float*)d_in[11];
    const float* ln2_g = (const float*)d_in[12];
    const float* ln2_b = (const float*)d_in[13];
    const float* tag   = (const float*)d_in[14];
    const float* rf1w  = (const float*)d_in[15];
    const float* rf1b  = (const float*)d_in[16];
    const float* rf2w  = (const float*)d_in[17];
    const float* rf2b  = (const float*)d_in[18];
    const float* rlng  = (const float*)d_in[19];
    const float* rlnb  = (const float*)d_in[20];
    const float* lmw   = (const float*)d_in[21];
    const float* lmb   = (const float*)d_in[22];
    float* outp = (float*)d_out;

    const long BSD  = (long)BB * SS * DM;
    const long BS3D = (long)BB * SS * 3 * DM;
    const long SC   = (long)BB * NH * SS * SS;
    const long BSF  = (long)BB * SS * FFD;
    const long VT   = (long)BB * NH * DH * SS;
    const long WQ  = (long)NEXP * NL * 3 * DM * DM;
    const long WO  = (long)NEXP * NL * DM * DM;
    const long W1  = (long)NEXP * NL * FFD * DM;
    const long W2  = (long)NEXP * NL * DM * FFD;
    const long WLM = (long)VOCAB * DM;

    __bf16* bh = (__bf16*)d_ws;
    __bf16* wq_bf  = bh; bh += WQ;
    __bf16* wo_bf  = bh; bh += WO;
    __bf16* w1_bf  = bh; bh += W1;
    __bf16* w2_bf  = bh; bh += W2;
    __bf16* wlm_bf = bh; bh += WLM;
    __bf16* reps_bf = bh; bh += BSD;
    __bf16* fin_bf  = bh; bh += BSD;
    __bf16* qkv_bf  = bh; bh += BS3D;
    __bf16* p_bf    = bh; bh += SC;
    __bf16* vt_bf   = bh; bh += VT;
    __bf16* obuf_bf = bh; bh += BSD;
    __bf16* h1_bf   = bh; bh += BSD;
    __bf16* ff_bf   = bh; bh += BSF;
    __bf16* xA_bf   = bh; bh += BSD;
    __bf16* xB_bf   = bh; bh += BSD;
    float* f = (float*)bh;
    float* reps   = f; f += BSD;
    float* scores = f; f += SC;
    float* proj   = f; f += BSD;
    float* h1     = f; f += BSD;
    float* xA     = f; f += BSD;
    float* xB     = f; f += BSD;
    float* summ   = f; f += BB * DM;
    float* rh     = f; f += BB * RHD;
    float* rlg    = f; f += 256;
    float* accv   = f; f += 8;
    int* visits   = (int*)f;
    int* active   = visits + BB * NEXP;
    int* contp    = active + BB;
    int* choicep  = contp + BB;
    int* cflag    = choicep + BB;

    const int eltBlocks = (int)((BSD + 255) / 256);   // 8192

    // weight preconversion fp32 -> bf16 (amortized across all GEMM reads)
    k_cvt<<<(int)((WQ  + 255) / 256), 256, 0, stream>>>(qkv_w, wq_bf, WQ);
    k_cvt<<<(int)((WO  + 255) / 256), 256, 0, stream>>>(out_w, wo_bf, WO);
    k_cvt<<<(int)((W1  + 255) / 256), 256, 0, stream>>>(ff1_w, w1_bf, W1);
    k_cvt<<<(int)((W2  + 255) / 256), 256, 0, stream>>>(ff2_w, w2_bf, W2);
    k_cvt<<<(int)((WLM + 255) / 256), 256, 0, stream>>>(lmw, wlm_bf, WLM);

    k_init<<<1, 128, 0, stream>>>(visits, active, accv);
    k_embed<<<eltBlocks, 256, 0, stream>>>(ids, emb, reps, reps_bf, fin_bf);

    for (int step = 0; step < 4; ++step) {
        k_summary<<<(BB * DM + 255) / 256, 256, 0, stream>>>(reps, summ);
        k_rfc1<<<(BB * RHD + 255) / 256, 256, 0, stream>>>(summ, rf1w, rf1b, rh);
        k_rfc2<<<1, 192, 0, stream>>>(rh, rf2w, rf2b, rlg);
        k_route<<<1, 32, 0, stream>>>(rlg, rlng, rlnb, visits, active, contp, choicep, cflag,
                                      accv, accv + 1);
        k_copyfinal<<<eltBlocks, 256, 0, stream>>>(reps_bf, fin_bf, cflag);

        const float*  xin    = reps;
        const __bf16* xin_bf = reps_bf;
        for (int layer = 0; layer < NL; ++layer) {
            float*  xout    = (layer == 0) ? xA : xB;
            __bf16* xout_bf = (layer == 0) ? xA_bf : xB_bf;
            // QKV: [S,512] x [1536,512]^T -> bf16
            k_bgemm<0, 0, 1><<<dim3(2, 24, BB), 128, 0, stream>>>(
                xin_bf, DM, (long)SS * DM,
                wq_bf, (long)3 * DM * DM, qkv_b, (long)3 * DM,
                nullptr, qkv_bf, 3 * DM, (long)SS * 3 * DM, DM, layer, choicep, contp);
            k_vt<<<(int)((VT + 255) / 256), 256, 0, stream>>>(qkv_bf, vt_bf, contp);
            // scores = Q K^T per (b,h)
            k_scores<<<dim3(16, 4, BB * NH), 32, 0, stream>>>(qkv_bf, scores, contp);
            k_softmax<<<BB * NH * SS, 256, 0, stream>>>(scores, p_bf, contp);
            // O = P V (via V^T rows)
            k_attnout<<<dim3(16, 1, BB * NH), 32, 0, stream>>>(p_bf, vt_bf, obuf_bf, contp);
            // proj: [S,512] x [512,512]^T -> fp32
            k_bgemm<0, 1, 0><<<dim3(2, 8, BB), 128, 0, stream>>>(
                obuf_bf, DM, (long)SS * DM,
                wo_bf, (long)DM * DM, out_b, (long)DM,
                proj, nullptr, DM, (long)SS * DM, DM, layer, choicep, contp);
            k_resid_ln<<<BB * SS, 256, 0, stream>>>(xin, proj, ln1_g, ln1_b, h1, h1_bf,
                                                    choicep, contp, layer);
            // FF1 (+ReLU): [S,512] x [2048,512]^T -> bf16
            k_bgemm<1, 0, 1><<<dim3(2, 32, BB), 128, 0, stream>>>(
                h1_bf, DM, (long)SS * DM,
                w1_bf, (long)FFD * DM, ff1_b, (long)FFD,
                nullptr, ff_bf, FFD, (long)SS * FFD, DM, layer, choicep, contp);
            // FF2: [S,2048] x [512,2048]^T -> fp32
            k_bgemm<0, 1, 0><<<dim3(2, 8, BB), 128, 0, stream>>>(
                ff_bf, FFD, (long)SS * FFD,
                w2_bf, (long)DM * FFD, ff2_b, (long)DM,
                proj, nullptr, DM, (long)SS * DM, FFD, layer, choicep, contp);
            k_resid_ln<<<BB * SS, 256, 0, stream>>>(h1, proj, ln2_g, ln2_b, xout, xout_bf,
                                                    choicep, contp, layer);
            xin = xout;
            xin_bf = xout_bf;
        }
        k_apply<<<eltBlocks, 256, 0, stream>>>(xB, tag, choicep, contp, reps, reps_bf);
    }

    k_copyfinal<<<eltBlocks, 256, 0, stream>>>(reps_bf, fin_bf, active);
    // LM head: [4096,512] x [32000,512]^T -> d_out
    k_bgemm<0, 1, 0><<<dim3(32, 500, 1), 128, 0, stream>>>(
        fin_bf, DM, 0L,
        wlm_bf, 0L, lmb, 0L,
        outp, nullptr, VOCAB, 0L, DM, 0, (const int*)nullptr, (const int*)nullptr);
    k_finalize<<<1, 1, 0, stream>>>(accv, outp + (long)BB * SS * VOCAB);
}
